// CausalSelfAttention_15513421873177
// MI455X (gfx1250) — compile-verified, matrix operand order checked by reading, not run on hardware
//
#include <hip/hip_runtime.h>
#include <hip/hip_bf16.h>

// ---------------------------------------------------------------------------
// CDNA5 (gfx1250) causal self-attention: bf16 WMMA GEMMs + wave32 flash attn
// ---------------------------------------------------------------------------

typedef __bf16 v16bf __attribute__((ext_vector_type(16)));
typedef float  v8f   __attribute__((ext_vector_type(8)));

union Frag {
    v16bf          v;
    uint4          q[2];
    unsigned short s[16];
};

__device__ __forceinline__ unsigned short f32_to_bf16(float f) {
    union { float f; unsigned int u; } c; c.f = f;
    unsigned int u = c.u;
    u += 0x7FFFu + ((u >> 16) & 1u);           // round-to-nearest-even
    return (unsigned short)(u >> 16);
}

// Load a 16x32 bf16 WMMA A/B fragment from row-major bf16 memory.
// Per ISA 7.12.2 (16-bit A-matrix 16x32): lane = {half = lane>>4, r = lane&15};
// elements j=0..7  hold K = k0 + half*8 + j          (contiguous 16B)
// elements j=8..15 hold K = k0 + 16 + half*8 + (j-8) (contiguous 16B)
__device__ __forceinline__ Frag load_frag(const unsigned short* __restrict__ base,
                                          int ld, int row0, int k0, int lane) {
    const int half = lane >> 4, r = lane & 15;
    Frag f;
    const unsigned short* p = base + (size_t)(row0 + r) * ld + k0 + half * 8;
    f.q[0] = *(const uint4*)(p);
    f.q[1] = *(const uint4*)(p + 16);
    return f;
}

__device__ __forceinline__ v8f wmma_bf16(const Frag& a, const Frag& b, v8f c) {
    return __builtin_amdgcn_wmma_f32_16x16x32_bf16(
        false, a.v, false, b.v, (short)0, c, false, false);
}

// ---------------------------------------------------------------------------
// Kernel 1: f32 -> bf16 convert (x)
// ---------------------------------------------------------------------------
__global__ void cvt_bf16_kernel(const float* __restrict__ in,
                                unsigned short* __restrict__ out, int n) {
    int i = blockIdx.x * blockDim.x + threadIdx.x;
    if (i < n) out[i] = f32_to_bf16(in[i]);
}

// ---------------------------------------------------------------------------
// Kernel 2: transpose + convert weights: out[n*K + k] = bf16(in[k*N + n])
// ---------------------------------------------------------------------------
__global__ void transpose_cvt_kernel(const float* __restrict__ in,
                                     unsigned short* __restrict__ out,
                                     int K, int N) {
    int i = blockIdx.x * blockDim.x + threadIdx.x;
    if (i >= K * N) return;
    int n = i / K, k = i % K;
    out[i] = f32_to_bf16(in[(size_t)k * N + n]);
}

// ---------------------------------------------------------------------------
// Kernel 3: bf16 WMMA GEMM. C[M x N] (f32) = A[M x K] (bf16) * Bt[N x K]^T.
// One wave owns a 32x64 output tile (8 accumulators): per 32-K step this is
// 12 b128 loads feeding 8 v_wmma (B fragments reused across both M halves),
// plus global_prefetch_b8 one K-step ahead.
// ---------------------------------------------------------------------------
__global__ void __launch_bounds__(128)
gemm_bf16_kernel(const unsigned short* __restrict__ A,
                 const unsigned short* __restrict__ Bt,
                 float* __restrict__ C, int M, int N, int K) {
    const int wave = (blockIdx.x * blockDim.x + threadIdx.x) >> 5;
    const int lane = threadIdx.x & 31;
    const int nt   = N >> 6;
    const int m0   = (wave / nt) << 5;   // 32 rows
    const int n0   = (wave % nt) << 6;   // 64 cols
    if (m0 >= M) return;

    v8f acc[2][4];
#pragma unroll
    for (int h = 0; h < 2; ++h)
#pragma unroll
        for (int c = 0; c < 4; ++c)
#pragma unroll
            for (int j = 0; j < 8; ++j) acc[h][c][j] = 0.f;

    const int r15 = lane & 15, half8 = (lane >> 4) * 8;

    for (int kb = 0; kb < K; kb += 32) {
        // Prefetch next K-step (lowered to global_prefetch_b8).
        if (kb + 32 < K) {
            __builtin_prefetch(A  + (size_t)(m0 + r15) * K + kb + 32 + half8, 0, 3);
            __builtin_prefetch(Bt + (size_t)(n0 + r15) * K + kb + 32 + half8, 0, 3);
        }
        Frag a0 = load_frag(A, K, m0,      kb, lane);
        Frag a1 = load_frag(A, K, m0 + 16, kb, lane);
#pragma unroll
        for (int c = 0; c < 4; ++c) {
            Frag b = load_frag(Bt, K, n0 + c * 16, kb, lane);
            acc[0][c] = wmma_bf16(a0, b, acc[0][c]);
            acc[1][c] = wmma_bf16(a1, b, acc[1][c]);
        }
    }

    // C/D layout: lane {half, n}; VGPR r holds row base + r + 8*half, col n0+c*16+n
    const int half = lane >> 4, n = lane & 15;
#pragma unroll
    for (int h = 0; h < 2; ++h)
#pragma unroll
        for (int c = 0; c < 4; ++c)
#pragma unroll
            for (int r = 0; r < 8; ++r)
                C[(size_t)(m0 + h * 16 + r + 8 * half) * N + n0 + c * 16 + n] =
                    acc[h][c][r];
}

// ---------------------------------------------------------------------------
// Kernel 4: split QKV + RoPE + convert to bf16.
// qkv f32 [2048][1536] -> q bf16 [16][2048][64], k bf16 [4][2048][64],
//                         v bf16 transposed [4][64][2048]
// ---------------------------------------------------------------------------
__global__ void rope_split_kernel(const float* __restrict__ qkv,
                                  unsigned short* __restrict__ qb,
                                  unsigned short* __restrict__ kb,
                                  unsigned short* __restrict__ vT) {
    int id = blockIdx.x * blockDim.x + threadIdx.x;
    const int TOT = 2048 * 24 * 32;
    if (id >= TOT) return;
    const int pair  = id & 31;        // rotation pair 0..31
    const int rest  = id >> 5;
    const int slotg = rest % 24;      // (group*6 + slot)
    const int t     = rest / 24;      // sequence position
    const int g     = slotg / 6, s = slotg % 6;

    const float* src = qkv + (size_t)t * 1536 + slotg * 64;
    const float e = src[2 * pair], o = src[2 * pair + 1];

    if (s < 5) {
        // RoPE: inv_freq = theta^(-2i/D), D=64, theta=1e4 (ln 1e4 = 9.2103404)
        const float invf = __expf(-((float)(2 * pair) * (1.0f / 64.0f)) *
                                  9.210340371976184f);
        const float ang = (float)t * invf;
        const float cs = __cosf(ang), sn = __sinf(ang);
        const unsigned short be = f32_to_bf16(e * cs - o * sn);
        const unsigned short bo = f32_to_bf16(o * cs + e * sn);
        if (s < 4) {
            unsigned short* dst = qb + ((size_t)(g * 4 + s) * 2048 + t) * 64 + 2 * pair;
            dst[0] = be; dst[1] = bo;
        } else {
            unsigned short* dst = kb + ((size_t)g * 2048 + t) * 64 + 2 * pair;
            dst[0] = be; dst[1] = bo;
        }
    } else {
        vT[((size_t)g * 64 + 2 * pair)     * 2048 + t] = f32_to_bf16(e);
        vT[((size_t)g * 64 + 2 * pair + 1) * 2048 + t] = f32_to_bf16(o);
    }
}

// ---------------------------------------------------------------------------
// Kernel 5: wave32 flash attention. One wave per (head, 16-query tile),
// 32 keys per iteration, online softmax, WMMA for QK^T and PV.
// P transposed C-layout -> A-layout via a private 2KB LDS slice.
// ---------------------------------------------------------------------------
__global__ void __launch_bounds__(64)
flash_attn_kernel(const unsigned short* __restrict__ qbuf,
                  const unsigned short* __restrict__ kbuf,
                  const unsigned short* __restrict__ vTb,
                  unsigned short* __restrict__ yb) {
    __shared__ float sP[2][16 * 32];

    const int lane = threadIdx.x & 31;
    const int wid  = threadIdx.x >> 5;
    const int wave = blockIdx.x * 2 + wid;
    const int qt   = wave & 127;          // 128 query tiles
    const int h    = wave >> 7;           // head 0..15
    const int g    = h >> 2;              // kv group
    const int q0   = qt << 4;

    const unsigned short* Q  = qbuf + (size_t)h * 2048 * 64;
    const unsigned short* Km = kbuf + (size_t)g * 2048 * 64;
    const unsigned short* Vt = vTb  + (size_t)g * 64 * 2048;
    float* myP = sP[wid];

    const int half = lane >> 4, n = lane & 15;
    const float scale = 0.125f;           // 1/sqrt(64)

    const Frag aq0 = load_frag(Q, 64, q0, 0,  lane);
    const Frag aq1 = load_frag(Q, 64, q0, 32, lane);

    v8f o[4];
    float mrow[8], lrow[8];
#pragma unroll
    for (int r = 0; r < 8; ++r) { mrow[r] = -1e30f; lrow[r] = 0.f; }
#pragma unroll
    for (int c = 0; c < 4; ++c)
#pragma unroll
        for (int j = 0; j < 8; ++j) o[c][j] = 0.f;

    for (int kb0 = 0; kb0 < q0 + 16; kb0 += 32) {
        // ---- S = Q K^T for 32 keys (two 16x16 tiles) ----
        v8f s0, s1;
#pragma unroll
        for (int j = 0; j < 8; ++j) { s0[j] = 0.f; s1[j] = 0.f; }
        {
            Frag b;
            b = load_frag(Km, 64, kb0,      0,  lane); s0 = wmma_bf16(aq0, b, s0);
            b = load_frag(Km, 64, kb0,      32, lane); s0 = wmma_bf16(aq1, b, s0);
            b = load_frag(Km, 64, kb0 + 16, 0,  lane); s1 = wmma_bf16(aq0, b, s1);
            b = load_frag(Km, 64, kb0 + 16, 32, lane); s1 = wmma_bf16(aq1, b, s1);
        }

        // ---- scale + causal mask + block row-max ----
        float bm[8];
#pragma unroll
        for (int r = 0; r < 8; ++r) {
            const int qi = q0 + r + 8 * half;
            float v0 = (kb0 + n      <= qi) ? s0[r] * scale : -1e30f;
            float v1 = (kb0 + 16 + n <= qi) ? s1[r] * scale : -1e30f;
            s0[r] = v0; s1[r] = v1;
            bm[r] = fmaxf(v0, v1);
        }
#pragma unroll
        for (int mk = 1; mk <= 8; mk <<= 1)
#pragma unroll
            for (int r = 0; r < 8; ++r)
                bm[r] = fmaxf(bm[r], __shfl_xor(bm[r], mk, 32));

        // ---- online softmax update ----
        float rs[8];
#pragma unroll
        for (int r = 0; r < 8; ++r) {
            const float mn = fmaxf(mrow[r], bm[r]);
            const float sc = __expf(mrow[r] - mn);
            mrow[r] = mn;
            lrow[r] *= sc;
#pragma unroll
            for (int c = 0; c < 4; ++c) o[c][r] *= sc;
            const float p0 = __expf(s0[r] - mn);
            const float p1 = __expf(s1[r] - mn);
            s0[r] = p0; s1[r] = p1;
            rs[r] = p0 + p1;
        }
#pragma unroll
        for (int mk = 1; mk <= 8; mk <<= 1)
#pragma unroll
            for (int r = 0; r < 8; ++r)
                rs[r] += __shfl_xor(rs[r], mk, 32);
#pragma unroll
        for (int r = 0; r < 8; ++r) lrow[r] += rs[r];

        // ---- transpose P (C-layout -> A-layout) through LDS ----
#pragma unroll
        for (int r = 0; r < 8; ++r) {
            const int row = r + 8 * half;
            myP[row * 32 + n]      = s0[r];
            myP[row * 32 + 16 + n] = s1[r];
        }
        __builtin_amdgcn_wave_barrier();
        asm volatile("s_wait_dscnt 0x0" ::: "memory");

        Frag pf;
        {
            const float* pr = myP + (lane & 15) * 32 + half * 8;
#pragma unroll
            for (int j = 0; j < 8; ++j) pf.s[j]     = f32_to_bf16(pr[j]);
#pragma unroll
            for (int j = 0; j < 8; ++j) pf.s[8 + j] = f32_to_bf16(pr[16 + j]);
        }
        __builtin_amdgcn_wave_barrier();
        asm volatile("" ::: "memory");

        // ---- O += P * V (V stored transposed [d][t]) ----
#pragma unroll
        for (int c = 0; c < 4; ++c) {
            Frag bv = load_frag(Vt, 2048, c * 16, kb0, lane);
            o[c] = wmma_bf16(pf, bv, o[c]);
        }
    }

    // ---- epilogue: normalize and store bf16 y[t][h*64 + d] ----
#pragma unroll
    for (int r = 0; r < 8; ++r) {
        const float inv = 1.0f / lrow[r];
#pragma unroll
        for (int c = 0; c < 4; ++c) {
            yb[(size_t)(q0 + r + 8 * half) * 1024 + h * 64 + c * 16 + n] =
                f32_to_bf16(o[c][r] * inv);
        }
    }
}

// ---------------------------------------------------------------------------
// Launch: x -> bf16; w^T -> bf16; QKV GEMM; RoPE/split; flash attn; out proj.
// ---------------------------------------------------------------------------
extern "C" void kernel_launch(void* const* d_in, const int* in_sizes, int n_in,
                              void* d_out, int out_size, void* d_ws, size_t ws_size,
                              hipStream_t stream) {
    const float* x      = (const float*)d_in[0];   // [2048][1024]
    const float* w_attn = (const float*)d_in[1];   // [1024][1536]
    const float* w_proj = (const float*)d_in[2];   // [1024][1024]
    float*       out    = (float*)d_out;           // [2048][1024]

    char* ws = (char*)d_ws;
    unsigned short* xb  = (unsigned short*)(ws);                 // 2048*1024 bf16
    unsigned short* waT = (unsigned short*)(ws + 4194304);       // 1536*1024 bf16
    unsigned short* wpT = (unsigned short*)(ws + 7340032);       // 1024*1024 bf16
    float*          qkv = (float*)(ws + 9437184);                // 2048*1536 f32
    unsigned short* qb  = (unsigned short*)(ws + 22020096);      // 16*2048*64 bf16
    unsigned short* kb  = (unsigned short*)(ws + 26214400);      // 4*2048*64 bf16
    unsigned short* vT  = (unsigned short*)(ws + 27262976);      // 4*64*2048 bf16
    unsigned short* yb  = (unsigned short*)(ws + 28311552);      // 2048*1024 bf16

    // 1. x -> bf16
    cvt_bf16_kernel<<<(2048 * 1024 + 255) / 256, 256, 0, stream>>>(x, xb, 2048 * 1024);
    // 2. weight transposes -> bf16 [N][K]
    transpose_cvt_kernel<<<(1024 * 1536 + 255) / 256, 256, 0, stream>>>(w_attn, waT, 1024, 1536);
    transpose_cvt_kernel<<<(1024 * 1024 + 255) / 256, 256, 0, stream>>>(w_proj, wpT, 1024, 1024);
    // 3. QKV GEMM: [2048x1536] = xb[2048x1024] * waT^T  (64 x 24 = 1536 waves)
    gemm_bf16_kernel<<<384, 128, 0, stream>>>(xb, waT, qkv, 2048, 1536, 1024);
    // 4. split + RoPE + bf16 (V transposed)
    rope_split_kernel<<<(2048 * 24 * 32 + 255) / 256, 256, 0, stream>>>(qkv, qb, kb, vT);
    // 5. flash attention: 16 heads * 128 q-tiles = 2048 waves, 2 waves/block
    flash_attn_kernel<<<1024, 64, 0, stream>>>(qb, kb, vT, yb);
    // 6. output projection: out[2048x1024] = yb * wpT^T  (64 x 16 = 1024 waves)
    gemm_bf16_kernel<<<256, 128, 0, stream>>>(yb, wpT, out, 2048, 1024, 1024);
}
